// CAGCL_40286793237099
// MI455X (gfx1250) — compile-verified, matrix operand order checked by reading, not run on hardware
//
#include <hip/hip_runtime.h>
#include <hip/hip_bf16.h>
#include <math.h>

#define NNODES 100000
#define NEDGES 1600000
#define DIM    128
#define NSUB   5000          // contrastive subsample count (step 20)
#define SSTEP  20

typedef __attribute__((ext_vector_type(16))) _Float16 v16h;
typedef __attribute__((ext_vector_type(8)))  float    v8f;
typedef __attribute__((ext_vector_type(8)))  _Float16 h8;

// ---------------------------------------------------------------------------
// WMMA fragment builders (wave32, V_WMMA_F32_16X16X32_F16 layout)
// A (16x32 f16): lane L -> row m=L&15; halves 0..7 = K[kb..kb+7],
//                halves 8..15 = K[kb+16..kb+23], kb=(L>>4)*8.
// B (32x16, from pre-transposed W^T [Nout,K]): same pattern with n=L&15.
// C/D (16x16 f32, 8 VGPRs): lane L -> col n=L&15, rows m=(L>>4)*8 + r.
// ---------------------------------------------------------------------------
__device__ inline v16h make_a_frag_f32(const float* rp) {
    const float4* p0 = (const float4*)(rp);
    const float4* p1 = (const float4*)(rp + 16);
    float4 a0 = p0[0], a1 = p0[1], a2 = p1[0], a3 = p1[1];
    v16h f;
    f[0]=(_Float16)a0.x; f[1]=(_Float16)a0.y; f[2]=(_Float16)a0.z; f[3]=(_Float16)a0.w;
    f[4]=(_Float16)a1.x; f[5]=(_Float16)a1.y; f[6]=(_Float16)a1.z; f[7]=(_Float16)a1.w;
    f[8]=(_Float16)a2.x; f[9]=(_Float16)a2.y; f[10]=(_Float16)a2.z; f[11]=(_Float16)a2.w;
    f[12]=(_Float16)a3.x; f[13]=(_Float16)a3.y; f[14]=(_Float16)a3.z; f[15]=(_Float16)a3.w;
    return f;
}

__device__ inline v16h make_b_frag_f16(const _Float16* rp) {
    h8 b0 = *(const h8*)(rp);
    h8 b1 = *(const h8*)(rp + 16);
    v16h f;
#pragma unroll
    for (int i = 0; i < 8; ++i) { f[i] = b0[i]; f[8 + i] = b1[i]; }
    return f;
}

__device__ inline v8f mm_acc(const float* A, const _Float16* BT, int K,
                             int mbase, int nbase, int lane, v8f acc) {
    const int m  = mbase + (lane & 15);
    const int n  = nbase + (lane & 15);
    const int kb = (lane >> 4) * 8;
    const float*    arow = A  + (size_t)m * K + kb;
    const _Float16* brow = BT + (size_t)n * K + kb;
    for (int k0 = 0; k0 < K; k0 += 32) {
        __builtin_prefetch(arow + k0 + 64, 0, 1);   // global_prefetch_b8
        v16h af = make_a_frag_f32(arow + k0);
        v16h bf = make_b_frag_f16(brow + k0);
        acc = __builtin_amdgcn_wmma_f32_16x16x32_f16(false, af, false, bf,
                                                     (short)0, acc, false, false);
    }
    return acc;
}

// C[M x Nout] = sum_s As @ Bs^T + bias, Bs pre-transposed f16 [Nout x K].
// lda == K (row-major, contiguous). ldc/col_off let us write column slices.
__global__ void k_wmma_gemm(const float* A0, const _Float16* B0,
                            const float* A1, const _Float16* B1,
                            const float* A2, const _Float16* B2,
                            const float* bias, float* C,
                            int M, int Nout, int K, int ldc, int col_off, int nsrc) {
    const int lane = threadIdx.x & 31;
    const int wave = blockIdx.x * (blockDim.x >> 5) + (threadIdx.x >> 5);
    const int ntiles = Nout >> 4;
    const int total  = (M >> 4) * ntiles;
    if (wave >= total) return;                 // wave-uniform: EXEC stays all-1s
    const int mbase = (wave / ntiles) << 4;
    const int nbase = (wave % ntiles) << 4;
    v8f acc = {};
    acc = mm_acc(A0, B0, K, mbase, nbase, lane, acc);
    if (nsrc > 1) acc = mm_acc(A1, B1, K, mbase, nbase, lane, acc);
    if (nsrc > 2) acc = mm_acc(A2, B2, K, mbase, nbase, lane, acc);
    const int nn = nbase + (lane & 15);
    const float bv = bias ? bias[nn] : 0.0f;
    const int m0 = mbase + ((lane >> 4) << 3);
#pragma unroll
    for (int r = 0; r < 8; ++r)
        C[(size_t)(m0 + r) * ldc + col_off + nn] = acc[r] + bv;
}

// ---------------------------------------------------------------------------
// Small helper kernels
// ---------------------------------------------------------------------------
__global__ void k_zero(float* p, long long n) {
    long long i = (long long)blockIdx.x * blockDim.x + threadIdx.x;
    if (i < n) p[i] = 0.0f;
}

// WT[n*K + k] = (f16) W[k*Nout + n]
__global__ void k_transpose_f16(const float* W, _Float16* WT, int K, int Nout) {
    int i = blockIdx.x * blockDim.x + threadIdx.x;
    if (i >= K * Nout) return;
    int k = i / Nout, n = i % Nout;
    WT[(size_t)n * K + k] = (_Float16)W[i];
}

// tiny-K dense: C[:, col_off:col_off+Nout] = A[MxK] @ W[KxNout] + b
__global__ void k_small_gemm(const float* A, const float* W, const float* b,
                             float* C, int M, int K, int Nout, int ldc, int col_off) {
    int i = blockIdx.x * blockDim.x + threadIdx.x;
    if (i >= M * Nout) return;
    int row = i / Nout, col = i % Nout;
    float acc = b[col];
    for (int k = 0; k < K; ++k) acc += A[(size_t)row * K + k] * W[k * Nout + col];
    C[(size_t)row * ldc + col_off + col] = acc;
}

// column stats (requires Dc divides blockDim.x)
__global__ void k_bn_stats(const float* X, float* sum, float* sumsq, int M, int Dc) {
    int tid = threadIdx.x;
    int rpb = blockDim.x / Dc;
    int c = tid % Dc;
    int r = blockIdx.x * rpb + tid / Dc;
    int stride = gridDim.x * rpb;
    float s = 0.0f, ss = 0.0f;
    for (; r < M; r += stride) {
        float v = X[(size_t)r * Dc + c];
        s += v; ss += v * v;
    }
    atomicAdd(&sum[c], s);
    atomicAdd(&sumsq[c], ss);
}

// act: 0 = none, 1 = leaky_relu(0.01), 2 = relu
__global__ void k_bn_apply(const float* X, float* Y, const float* sum,
                           const float* sumsq, int M, int Dc, int act) {
    long long i = (long long)blockIdx.x * blockDim.x + threadIdx.x;
    if (i >= (long long)M * Dc) return;
    int c = (int)(i % Dc);
    float mean = sum[c] / (float)M;
    float var  = sumsq[c] / (float)M - mean * mean;
    float v = (X[i] - mean) * rsqrtf(var + 1e-5f);
    if (act == 1) v = (v > 0.0f) ? v : 0.01f * v;
    else if (act == 2) v = fmaxf(v, 0.0f);
    Y[i] = v;
}

__global__ void k_edge_count(const int* dst, const int* et, float* cnt, int E, int N) {
    int e = blockIdx.x * blockDim.x + threadIdx.x;
    if (e >= E) return;
    atomicAdd(&cnt[(size_t)et[e] * N + dst[e]], 1.0f);
}

// per (edge, channel): agg[r][dst][c] += x[src][c]
__global__ void k_edge_scatter(const float* x, const int* src, const int* dst,
                               const int* et, float* agg, long long total, int N) {
    long long i = (long long)blockIdx.x * blockDim.x + threadIdx.x;
    if (i >= total) return;
    int e = (int)(i >> 7);
    int c = (int)(i & 127);
    int r = et[e];
    atomicAdd(&agg[(((size_t)r * N + dst[e]) << 7) + c],
              x[((size_t)src[e] << 7) + c]);
}

__global__ void k_agg_div(float* agg, const float* cnt, int N) {
    long long i = (long long)blockIdx.x * blockDim.x + threadIdx.x;
    if (i >= (long long)2 * N * DIM) return;
    int r = (int)(i / ((long long)N * DIM));
    int node = (int)((i >> 7) % N);
    agg[i] /= fmaxf(cnt[(size_t)r * N + node], 1.0f);
}

// x1[dst][c] += 0.1 * w * x[src][c] for edges with w > 0.5
__global__ void k_enhance(const float* x, const int* src, const int* dst,
                          const float* w, float* x1, long long total) {
    long long i = (long long)blockIdx.x * blockDim.x + threadIdx.x;
    if (i >= total) return;
    int e = (int)(i >> 7);
    float we = w[e];
    if (we <= 0.5f) return;
    int c = (int)(i & 127);
    atomicAdd(&x1[((size_t)dst[e] << 7) + c], 0.1f * we * x[((size_t)src[e] << 7) + c]);
}

// logits = feat @ W_o2 + b ; out = log_softmax
__global__ void k_logits(const float* feat, const float* W, const float* b,
                         float* out, int M) {
    int i = blockIdx.x * blockDim.x + threadIdx.x;
    if (i >= M) return;
    const float* f = feat + ((size_t)i << 7);
    float l0 = b[0], l1 = b[1];
#pragma unroll 8
    for (int k = 0; k < DIM; ++k) { float v = f[k]; l0 += v * W[2 * k]; l1 += v * W[2 * k + 1]; }
    float m = fmaxf(l0, l1);
    float lse = m + logf(expf(l0 - m) + expf(l1 - m));
    out[2 * i] = l0 - lse;
    out[2 * i + 1] = l1 - lse;
}

__global__ void k_l2norm(float* p, int M, int Dc) {
    int i = blockIdx.x * blockDim.x + threadIdx.x;
    if (i >= M) return;
    float* row = p + (size_t)i * Dc;
    float sq = 0.0f;
    for (int k = 0; k < Dc; ++k) sq += row[k] * row[k];
    float inv = 1.0f / fmaxf(sqrtf(sq), 1e-12f);
    for (int k = 0; k < Dc; ++k) row[k] *= inv;
}

// fused contrastive: one block per subsampled row i, never materialize sim.
// scal[0] += row loss; scal[1] += sum(sim*cmask) (for nsl); scal[2] += ||f_i||
__global__ void k_loss(const float* proj, const float* comm, float* scal) {
    __shared__ float fi[64];
    __shared__ float s0[256], s1[256], s2[256];
    const int i = blockIdx.x;
    const int tid = threadIdx.x;
    const int row = i * SSTEP;
    if (tid < 64) fi[tid] = proj[((size_t)row << 6) + tid];
    const float c0 = comm[2 * row], c1 = comm[2 * row + 1];
    __syncthreads();
    float possum = 0.0f, poscnt = 0.0f, negsum = 0.0f;
    for (int j = tid; j < NSUB; j += 256) {
        const float* fj = proj + ((size_t)(j * SSTEP) << 6);
        float s = 0.0f;
#pragma unroll 8
        for (int k = 0; k < 64; ++k) s += fi[k] * fj[k];
        s *= 10.0f;                       // 1 / TEMP
        float cs = c0 * comm[2 * j * SSTEP] + c1 * comm[2 * j * SSTEP + 1];
        if (j != i) {
            if (cs > 0.0f) { possum += s; poscnt += 1.0f; }
            else           { negsum += expf(s); }
        }
    }
    s0[tid] = possum; s1[tid] = poscnt; s2[tid] = negsum;
    __syncthreads();
    for (int off = 128; off > 0; off >>= 1) {
        if (tid < off) { s0[tid] += s0[tid + off]; s1[tid] += s1[tid + off]; s2[tid] += s2[tid + off]; }
        __syncthreads();
    }
    if (tid == 0) {
        float pos = s0[0] / fmaxf(s1[0], 1.0f);
        atomicAdd(&scal[0], -pos + logf(s2[0] + 1e-10f));
        atomicAdd(&scal[1], s0[0]);
        float sq = 0.0f;
        for (int k = 0; k < 64; ++k) sq += fi[k] * fi[k];
        atomicAdd(&scal[2], sqrtf(sq));
    }
}

__global__ void k_finalize(const float* scal, float* out) {
    if (blockIdx.x == 0 && threadIdx.x == 0) {
        float loss = scal[0] / (float)NSUB;
        float nsl  = -scal[1] / ((float)NSUB * (float)NSUB);
        float l2   = scal[2] / (float)NSUB;
        out[0] = 0.3f * (loss + 0.2f * nsl + 0.001f * l2);   // CW*(loss + NSW*nsl + 1e-3*l2)
    }
}

// ---------------------------------------------------------------------------
// Host orchestration
// ---------------------------------------------------------------------------
static inline char* carve(char*& p, size_t bytes) {
    char* r = p;
    p += (bytes + 255) & ~(size_t)255;
    return r;
}
static inline unsigned gsz(long long n) { return (unsigned)((n + 255) / 256); }

extern "C" void kernel_launch(void* const* d_in, const int* in_sizes, int n_in,
                              void* d_out, int out_size, void* d_ws, size_t ws_size,
                              hipStream_t stream) {
    (void)in_sizes; (void)n_in; (void)out_size; (void)ws_size;
    const int N = NNODES, E = NEDGES;

    const float* tweet   = (const float*)d_in[0];
    const float* numP    = (const float*)d_in[1];
    const float* catP    = (const float*)d_in[2];
    const float* commE   = (const float*)d_in[3];
    const float* ecw     = (const float*)d_in[4];
    const int*   eidx    = (const int*)d_in[5];
    const int*   etype   = (const int*)d_in[6];
    const float *W_tw=(const float*)d_in[7],  *b_tw=(const float*)d_in[8];
    const float *W_nu=(const float*)d_in[9],  *b_nu=(const float*)d_in[10];
    const float *W_ca=(const float*)d_in[11], *b_ca=(const float*)d_in[12];
    const float *W_co=(const float*)d_in[13], *b_co=(const float*)d_in[14];
    const float *W_i1=(const float*)d_in[15], *b_i1=(const float*)d_in[16];
    const float *W_i2=(const float*)d_in[17], *b_i2=(const float*)d_in[18];
    const float *Wrel1=(const float*)d_in[19], *Wroot1=(const float*)d_in[20], *broot1=(const float*)d_in[21];
    const float *Wrel2=(const float*)d_in[22], *Wroot2=(const float*)d_in[23], *broot2=(const float*)d_in[24];
    const float *W_o1=(const float*)d_in[25], *b_o1=(const float*)d_in[26];
    const float *W_o2=(const float*)d_in[27], *b_o2=(const float*)d_in[28];
    const float *W_p1=(const float*)d_in[29], *b_p1=(const float*)d_in[30];
    const float *W_p2=(const float*)d_in[31], *b_p2=(const float*)d_in[32];
    const int* esrc = eidx;
    const int* edst = eidx + E;
    float* out = (float*)d_out;

    // workspace carve
    char* p = (char*)d_ws;
    _Float16* wtTW = (_Float16*)carve(p, (size_t)32 * 768 * 2);
    _Float16* wtI1 = (_Float16*)carve(p, (size_t)128 * 128 * 2);
    _Float16* wtI2 = (_Float16*)carve(p, (size_t)128 * 128 * 2);
    _Float16* wtR1 = (_Float16*)carve(p, (size_t)128 * 128 * 2);   // Wroot1^T
    _Float16* wtA1 = (_Float16*)carve(p, (size_t)128 * 128 * 2);   // Wrel1[0]^T
    _Float16* wtB1 = (_Float16*)carve(p, (size_t)128 * 128 * 2);   // Wrel1[1]^T
    _Float16* wtR2 = (_Float16*)carve(p, (size_t)128 * 128 * 2);
    _Float16* wtA2 = (_Float16*)carve(p, (size_t)128 * 128 * 2);
    _Float16* wtB2 = (_Float16*)carve(p, (size_t)128 * 128 * 2);
    _Float16* wtO1 = (_Float16*)carve(p, (size_t)128 * 128 * 2);
    _Float16* wtP1 = (_Float16*)carve(p, (size_t)128 * 128 * 2);
    _Float16* wtP2 = (_Float16*)carve(p, (size_t)64 * 128 * 2);
    float* x    = (float*)carve(p, (size_t)N * DIM * 4);
    float* y    = (float*)carve(p, (size_t)N * DIM * 4);
    float* x1   = (float*)carve(p, (size_t)N * DIM * 4);
    float* feat = (float*)carve(p, (size_t)N * DIM * 4);
    float* proj = (float*)carve(p, (size_t)N * 64 * 4);
    float* agg  = (float*)carve(p, (size_t)2 * N * DIM * 4);
    float* cnt  = (float*)carve(p, (size_t)2 * N * 4);
    float* bnst = (float*)carve(p, 256 * 4);          // [0..127]=sum [128..255]=sumsq
    float* scal = (float*)carve(p, 16 * 4);
    float* bsum = bnst, *bsq = bnst + 128;

    const long long ND = (long long)N * DIM;
    const long long EC = (long long)E * DIM;

    // weight transpose/convert to f16 (tiny)
    k_transpose_f16<<<gsz(768 * 32), 256, 0, stream>>>(W_tw, wtTW, 768, 32);
    k_transpose_f16<<<gsz(128 * 128), 256, 0, stream>>>(W_i1, wtI1, 128, 128);
    k_transpose_f16<<<gsz(128 * 128), 256, 0, stream>>>(W_i2, wtI2, 128, 128);
    k_transpose_f16<<<gsz(128 * 128), 256, 0, stream>>>(Wroot1, wtR1, 128, 128);
    k_transpose_f16<<<gsz(128 * 128), 256, 0, stream>>>(Wrel1, wtA1, 128, 128);
    k_transpose_f16<<<gsz(128 * 128), 256, 0, stream>>>(Wrel1 + 128 * 128, wtB1, 128, 128);
    k_transpose_f16<<<gsz(128 * 128), 256, 0, stream>>>(Wroot2, wtR2, 128, 128);
    k_transpose_f16<<<gsz(128 * 128), 256, 0, stream>>>(Wrel2, wtA2, 128, 128);
    k_transpose_f16<<<gsz(128 * 128), 256, 0, stream>>>(Wrel2 + 128 * 128, wtB2, 128, 128);
    k_transpose_f16<<<gsz(128 * 128), 256, 0, stream>>>(W_o1, wtO1, 128, 128);
    k_transpose_f16<<<gsz(128 * 128), 256, 0, stream>>>(W_p1, wtP1, 128, 128);
    k_transpose_f16<<<gsz(128 * 64), 256, 0, stream>>>(W_p2, wtP2, 128, 64);

    unsigned gemm32 = (unsigned)(((N / 16) * 2 + 7) / 8);    // Nout=32
    unsigned gemm128 = (unsigned)(((N / 16) * 8 + 7) / 8);   // Nout=128
    unsigned gemm64 = (unsigned)(((N / 16) * 4 + 7) / 8);    // Nout=64

    // --- feature blocks -> pre-BN concat in y ---
    k_wmma_gemm<<<gemm32, 256, 0, stream>>>(tweet, wtTW, nullptr, nullptr, nullptr, nullptr,
                                            b_tw, y, N, 32, 768, DIM, 0, 1);
    k_small_gemm<<<gsz((long long)N * 32), 256, 0, stream>>>(numP, W_nu, b_nu, y, N, 10, 32, DIM, 32);
    k_small_gemm<<<gsz((long long)N * 32), 256, 0, stream>>>(catP, W_ca, b_ca, y, N, 10, 32, DIM, 64);
    k_small_gemm<<<gsz((long long)N * 32), 256, 0, stream>>>(commE, W_co, b_co, y, N, 2, 32, DIM, 96);
    k_zero<<<1, 256, 0, stream>>>(bnst, 256);
    k_bn_stats<<<1024, 256, 0, stream>>>(y, bsum, bsq, N, DIM);
    k_bn_apply<<<gsz(ND), 256, 0, stream>>>(y, x, bsum, bsq, N, DIM, 1);

    // --- input blocks 1 & 2 ---
    k_wmma_gemm<<<gemm128, 256, 0, stream>>>(x, wtI1, nullptr, nullptr, nullptr, nullptr,
                                             b_i1, y, N, DIM, DIM, DIM, 0, 1);
    k_zero<<<1, 256, 0, stream>>>(bnst, 256);
    k_bn_stats<<<1024, 256, 0, stream>>>(y, bsum, bsq, N, DIM);
    k_bn_apply<<<gsz(ND), 256, 0, stream>>>(y, x, bsum, bsq, N, DIM, 1);

    k_wmma_gemm<<<gemm128, 256, 0, stream>>>(x, wtI2, nullptr, nullptr, nullptr, nullptr,
                                             b_i2, y, N, DIM, DIM, DIM, 0, 1);
    k_zero<<<1, 256, 0, stream>>>(bnst, 256);
    k_bn_stats<<<1024, 256, 0, stream>>>(y, bsum, bsq, N, DIM);
    k_bn_apply<<<gsz(ND), 256, 0, stream>>>(y, x, bsum, bsq, N, DIM, 1);

    // --- RGCN layer 1 ---
    k_zero<<<gsz(2 * ND), 256, 0, stream>>>(agg, 2 * ND);
    k_zero<<<gsz(2 * N), 256, 0, stream>>>(cnt, 2 * N);
    k_edge_count<<<gsz(E), 256, 0, stream>>>(edst, etype, cnt, E, N);
    k_edge_scatter<<<gsz(EC), 256, 0, stream>>>(x, esrc, edst, etype, agg, EC, N);
    k_agg_div<<<gsz(2 * ND), 256, 0, stream>>>(agg, cnt, N);
    k_wmma_gemm<<<gemm128, 256, 0, stream>>>(x, wtR1, agg, wtA1, agg + ND, wtB1,
                                             broot1, y, N, DIM, DIM, DIM, 0, 3);
    k_zero<<<1, 256, 0, stream>>>(bnst, 256);
    k_bn_stats<<<1024, 256, 0, stream>>>(y, bsum, bsq, N, DIM);
    k_bn_apply<<<gsz(ND), 256, 0, stream>>>(y, x1, bsum, bsq, N, DIM, 0);
    k_enhance<<<gsz(EC), 256, 0, stream>>>(x, esrc, edst, ecw, x1, EC);

    // --- RGCN layer 2 (result x2 stored into x) ---
    k_zero<<<gsz(2 * ND), 256, 0, stream>>>(agg, 2 * ND);
    k_zero<<<gsz(2 * N), 256, 0, stream>>>(cnt, 2 * N);
    k_edge_count<<<gsz(E), 256, 0, stream>>>(edst, etype, cnt, E, N);
    k_edge_scatter<<<gsz(EC), 256, 0, stream>>>(x1, esrc, edst, etype, agg, EC, N);
    k_agg_div<<<gsz(2 * ND), 256, 0, stream>>>(agg, cnt, N);
    k_wmma_gemm<<<gemm128, 256, 0, stream>>>(x1, wtR2, agg, wtA2, agg + ND, wtB2,
                                             broot2, y, N, DIM, DIM, DIM, 0, 3);
    k_zero<<<1, 256, 0, stream>>>(bnst, 256);
    k_bn_stats<<<1024, 256, 0, stream>>>(y, bsum, bsq, N, DIM);
    k_bn_apply<<<gsz(ND), 256, 0, stream>>>(y, x, bsum, bsq, N, DIM, 0);

    // --- output block + classifier ---
    k_wmma_gemm<<<gemm128, 256, 0, stream>>>(x, wtO1, nullptr, nullptr, nullptr, nullptr,
                                             b_o1, y, N, DIM, DIM, DIM, 0, 1);
    k_zero<<<1, 256, 0, stream>>>(bnst, 256);
    k_bn_stats<<<1024, 256, 0, stream>>>(y, bsum, bsq, N, DIM);
    k_bn_apply<<<gsz(ND), 256, 0, stream>>>(y, feat, bsum, bsq, N, DIM, 1);
    k_logits<<<gsz(N), 256, 0, stream>>>(feat, W_o2, b_o2, out, N);

    // --- projection head (hidden h reuses x1) ---
    k_wmma_gemm<<<gemm128, 256, 0, stream>>>(feat, wtP1, nullptr, nullptr, nullptr, nullptr,
                                             b_p1, y, N, DIM, DIM, DIM, 0, 1);
    k_zero<<<1, 256, 0, stream>>>(bnst, 256);
    k_bn_stats<<<1024, 256, 0, stream>>>(y, bsum, bsq, N, DIM);
    k_bn_apply<<<gsz(ND), 256, 0, stream>>>(y, x1, bsum, bsq, N, DIM, 2);   // ReLU

    k_wmma_gemm<<<gemm64, 256, 0, stream>>>(x1, wtP2, nullptr, nullptr, nullptr, nullptr,
                                            b_p2, y, N, 64, DIM, 64, 0, 1);
    k_zero<<<1, 256, 0, stream>>>(bnst, 256);
    k_bn_stats<<<1024, 256, 0, stream>>>(y, bsum, bsq, N, 64);
    k_bn_apply<<<gsz((long long)N * 64), 256, 0, stream>>>(y, proj, bsum, bsq, N, 64, 0);
    k_l2norm<<<gsz(N), 256, 0, stream>>>(proj, N, 64);

    // --- contrastive loss ---
    k_zero<<<1, 256, 0, stream>>>(scal, 16);
    k_loss<<<NSUB, 256, 0, stream>>>(proj, commE, scal);
    k_finalize<<<1, 64, 0, stream>>>(scal, out + 2 * (size_t)N);
}
